// AdjLoss_10368051053022
// MI455X (gfx1250) — matrix operand order repaired and checked by reading, not compiled, not run on hardware
//
#include <hip/hip_runtime.h>
#include <hip/hip_bf16.h>

#define NN 8192
#define DD 256
#define TILES (NN / 16)   // 512 tiles per dimension

typedef __attribute__((ext_vector_type(16))) __bf16 v16bf;
typedef __attribute__((ext_vector_type(8)))  __bf16 v8bf;
typedef __attribute__((ext_vector_type(8)))  float  v8f;

// ---------------- workspace layout ----------------
// [0,            8 MB)  : adjacency bitmask, NN*NN bits (uint32 words)
// [8 MB,        12 MB)  : Xhi  bf16  (NN*DD)
// [12 MB,       16 MB)  : Xlo  bf16  (NN*DD)
// [16 MB,   16 MB+8 B)  : double accumulator
static constexpr size_t BITS_BYTES = (size_t)NN * NN / 8;       // 8 MB
static constexpr size_t XHALF_BYTES = (size_t)NN * DD * 2;      // 4 MB

__global__ void zero_ws_kernel(unsigned* __restrict__ bits, double* __restrict__ acc,
                               int nwords) {
  int idx = blockIdx.x * blockDim.x + threadIdx.x;
  int stride = gridDim.x * blockDim.x;
  for (int i = idx; i < nwords; i += stride) bits[i] = 0u;
  if (idx == 0) *acc = 0.0;
}

// fp32 -> (hi, lo) bf16 split: x ~= hi + lo
__global__ void cvt_kernel(const float* __restrict__ x, __bf16* __restrict__ hi,
                           __bf16* __restrict__ lo, int n) {
  int i = blockIdx.x * blockDim.x + threadIdx.x;
  if (i < n) {
    float v = x[i];
    __bf16 h = (__bf16)v;
    hi[i] = h;
    lo[i] = (__bf16)(v - (float)h);
  }
}

// Symmetric adjacency bitmask (duplicate edges collapse, like .at[].set(1))
__global__ void edge_kernel(const int* __restrict__ ei, unsigned* __restrict__ bits,
                            int e) {
  int t = blockIdx.x * blockDim.x + threadIdx.x;
  if (t < e) {
    unsigned s = (unsigned)ei[t];
    unsigned d = (unsigned)ei[e + t];
    unsigned b0 = s * (unsigned)NN + d;
    unsigned b1 = d * (unsigned)NN + s;
    atomicOr(&bits[b0 >> 5], 1u << (b0 & 31u));
    atomicOr(&bits[b1 >> 5], 1u << (b1 & 31u));
  }
}

__device__ __forceinline__ float bce_term(float z, bool adj) {
  // adj==0: min(softplus(z),100); adj==1: min(softplus(-z),100)
  float s = adj ? -z : z;
  float sp = fmaxf(s, 0.0f) + log1pf(__expf(-fabsf(s)));
  return fminf(sp, 100.0f);
}

// One wave per 16x16 output tile; block = 8 waves = 8 consecutive tj tiles.
// Only tiles with tj >= ti are computed (symmetry); off-diagonal weight = 2.
__global__ __launch_bounds__(256) void gram_bce_kernel(
    const __bf16* __restrict__ xhi, const __bf16* __restrict__ xlo,
    const unsigned* __restrict__ bits, double* __restrict__ acc) {
  const int lane = threadIdx.x & 31;
  const int wave = threadIdx.x >> 5;
  const int ti = blockIdx.y;
  const int tj = blockIdx.x * 8 + wave;
  if (tj < ti) return;  // lower-triangle tile: mirrored by weight-2 tiles

  const int half   = lane >> 4;   // 0: lanes 0-15, 1: lanes 16-31
  const int lane15 = lane & 15;

  // A-matrix (16x32 bf16): lane row = ti*16+lane15; per K-step this lane holds
  // K = [aoff, aoff+8) and [aoff+16, aoff+24) where aoff = half*8.
  const size_t arow_base = (size_t)(ti * 16 + lane15) * DD + half * 8;
  // B-matrix (32x16 bf16): column = row (tj*16+lane15) of X; this lane holds
  // K = [boff, boff+8) and [boff+16, boff+24) where boff = half*8.
  const size_t brow_base = (size_t)(tj * 16 + lane15) * DD + half * 8;

  v8f c = {};
#pragma unroll
  for (int k0 = 0; k0 < DD; k0 += 32) {
    const __bf16* pah = xhi + arow_base + k0;
    const __bf16* pal = xlo + arow_base + k0;
    v16bf ahi = __builtin_shufflevector(*(const v8bf*)pah, *(const v8bf*)(pah + 16),
                                        0, 1, 2, 3, 4, 5, 6, 7,
                                        8, 9, 10, 11, 12, 13, 14, 15);
    v16bf alo = __builtin_shufflevector(*(const v8bf*)pal, *(const v8bf*)(pal + 16),
                                        0, 1, 2, 3, 4, 5, 6, 7,
                                        8, 9, 10, 11, 12, 13, 14, 15);
    v16bf bhi = __builtin_shufflevector(*(const v8bf*)(xhi + brow_base + k0), *(const v8bf*)(xhi + brow_base + k0 + 16),
                                        0, 1, 2, 3, 4, 5, 6, 7,
                                        8, 9, 10, 11, 12, 13, 14, 15);
    v16bf blo = __builtin_shufflevector(*(const v8bf*)(xlo + brow_base + k0), *(const v8bf*)(xlo + brow_base + k0 + 16),
                                        0, 1, 2, 3, 4, 5, 6, 7,
                                        8, 9, 10, 11, 12, 13, 14, 15);
    // z = hi*hi + hi*lo + lo*hi  (~fp32 accuracy via bf16 WMMA)
    c = __builtin_amdgcn_wmma_f32_16x16x32_bf16(false, ahi, false, bhi, (short)0, c,
                                                false, false);
    c = __builtin_amdgcn_wmma_f32_16x16x32_bf16(false, ahi, false, blo, (short)0, c,
                                                false, false);
    c = __builtin_amdgcn_wmma_f32_16x16x32_bf16(false, alo, false, bhi, (short)0, c,
                                                false, false);
  }

  // Epilogue: C/D layout — lane holds N = lane15, M = v + half*8 for VGPR v.
  const float w = (ti == tj) ? 1.0f : 2.0f;
  const int j = tj * 16 + lane15;
  const int mbase = ti * 16 + half * 8;
  float lsum = 0.0f;
#pragma unroll
  for (int v = 0; v < 8; ++v) {
    int i = mbase + v;
    float z = c[v];
    unsigned bidx = (unsigned)i * (unsigned)NN + (unsigned)j;
    bool adj = (bits[bidx >> 5] >> (bidx & 31u)) & 1u;
    float t;
    if (i == j) {
      t = adj ? 100.0f : 0.0f;  // res==0 on diagonal -> log clamp
    } else {
      t = bce_term(z, adj);
    }
    lsum += w * t;
  }

  // wave32 butterfly reduction, then one f64 atomic per wave
  for (int m = 16; m > 0; m >>= 1) lsum += __shfl_xor(lsum, m, 32);
  if (lane == 0) atomicAdd(acc, (double)lsum);
}

__global__ void finalize_kernel(const double* __restrict__ acc, float* __restrict__ out) {
  if (threadIdx.x == 0 && blockIdx.x == 0) {
    out[0] = (float)(*acc / ((double)NN * (double)NN));
  }
}

extern "C" void kernel_launch(void* const* d_in, const int* in_sizes, int n_in,
                              void* d_out, int out_size, void* d_ws, size_t ws_size,
                              hipStream_t stream) {
  const float* l_enc = (const float*)d_in[0];
  const int* edge_index = (const int*)d_in[1];
  const int E = in_sizes[1] / 2;

  char* ws = (char*)d_ws;
  unsigned* bits = (unsigned*)ws;
  __bf16* xhi = (__bf16*)(ws + BITS_BYTES);
  __bf16* xlo = (__bf16*)(ws + BITS_BYTES + XHALF_BYTES);
  double* acc = (double*)(ws + BITS_BYTES + 2 * XHALF_BYTES);

  const int nwords = NN * NN / 32;  // 2M words
  zero_ws_kernel<<<2048, 256, 0, stream>>>(bits, acc, nwords);
  cvt_kernel<<<(NN * DD + 255) / 256, 256, 0, stream>>>(l_enc, xhi, xlo, NN * DD);
  edge_kernel<<<(E + 255) / 256, 256, 0, stream>>>(edge_index, bits, E);

  dim3 grid(TILES / 8, TILES);  // 64 x 512 blocks, 8 waves (tiles) per block
  gram_bce_kernel<<<grid, 256, 0, stream>>>(xhi, xlo, bits, acc);

  finalize_kernel<<<1, 64, 0, stream>>>(acc, (float*)d_out);
}
